// DeformableTransformerDecoderLayer_2465311228199
// MI455X (gfx1250) — matrix operand order repaired and checked by reading, not compiled, not run on hardware
//
#include <hip/hip_runtime.h>
#include <stdint.h>

// ---------------------------------------------------------------------------
// DeformableTransformerDecoderLayer for MI455X (gfx1250, wave32, WMMA).
// GEMMs: bf16 A/B, f32 accumulate via v_wmma_f32_16x16x32_bf16.
// Weight panels staged into LDS with global_load_async_to_lds_b128
// (ASYNCcnt) and shared by the 8 waves (8 M-tiles) of each block.
// ---------------------------------------------------------------------------

typedef __attribute__((ext_vector_type(16))) __bf16 v16bf;
typedef __attribute__((ext_vector_type(8)))  __bf16 v8bf;
typedef __attribute__((ext_vector_type(8)))  float  v8f;

#define D_MODEL 256
#define DFF_    1024
#define N_BATCH 8
#define LQ_     1024
#define NH_     8
#define NL_     4
#define NP_     4
#define HD_     32
#define LIN_    43520

// ---------------------------------------------------------------------------
// Weight convert + transpose:  W[K,Nc] f32  ->  WT[Nc,K] bf16
// ---------------------------------------------------------------------------
__global__ __launch_bounds__(256)
void wcvt_kernel(const float* __restrict__ W, __bf16* __restrict__ WT,
                 int K, int Nc)
{
    int idx = blockIdx.x * 256 + threadIdx.x;
    if (idx >= K * Nc) return;
    int k = idx / Nc;
    int c = idx - k * Nc;
    WT[(size_t)c * K + k] = (__bf16)W[idx];
}

// ---------------------------------------------------------------------------
// Elementwise add: q = tgt + query_pos
// ---------------------------------------------------------------------------
__global__ __launch_bounds__(256)
void addvec_kernel(const float* __restrict__ a, const float* __restrict__ b,
                   float* __restrict__ o, int n)
{
    int i = blockIdx.x * 256 + threadIdx.x;
    if (i < n) o[i] = a[i] + b[i];
}

// ---------------------------------------------------------------------------
// WMMA GEMM: out[M,Nc] = A[M,K] * WT[Nc,K](bf16 pre-transposed) + bias.
// Block = 8 waves = 8 consecutive 16-row M tiles x one 64-column group.
// K in chunks of 64: the block cooperatively stages the 64x64 bf16 B panel
// chunk into LDS with global_load_async_to_lds_b128 (each thread moves 32B,
// both global and LDS addresses advanced by the inst offset), waits on
// ASYNCcnt, barriers, then each wave runs 2 k-steps x 4 WMMA tiles with the
// B fragments served from LDS (ds_load_b128).
// A-fragment (16x32 bf16): lane L -> row L%16; lanes 0-15 hold K {0..15},
// lanes 16-31 hold K {16..31} (matches B fragment order).
// C/D: VGPR r -> row r + 8*(lane>=16), lane%16 -> column.
// Requires: M % 128 == 0, Nc % 64 == 0, K % 64 == 0 (true for all calls).
// ---------------------------------------------------------------------------
template<bool A_BF16, bool OUT_BF16, bool RELU, bool MASK>
__global__ __launch_bounds__(256)
void gemm_wmma_kernel(const void* __restrict__ Av,
                      const __bf16* __restrict__ WT,
                      const float* __restrict__ bias,
                      void* __restrict__ outv,
                      const unsigned char* __restrict__ rowMask,
                      int M, int Nc, int K)
{
    __shared__ __attribute__((aligned(32))) __bf16 sB[64 * 64];  // 8KB panel

    const int tid  = threadIdx.x;
    const int wid  = tid >> 5;
    const int lane = tid & 31;
    const int numCG = Nc >> 6;
    const int mg = blockIdx.x / numCG;        // 128-row group
    const int cg = blockIdx.x - mg * numCG;
    const int colBase = cg * 64;
    const int tileM = mg * 8 + wid;           // this wave's 16-row tile

    const int lm = lane & 15;
    const int hi = lane >> 4;
    const int kA0 = hi ? 16 : 0;
    const int kA1 = hi ? 24 : 8;
    const int kB  = hi ? 16 : 0;

    // cooperative staging assignment: thread -> (col, 16-K segment)
    const int scol = tid >> 2;                // 0..63
    const int skp  = (tid & 3) * 16;          // 0,16,32,48
    const __bf16* gB = WT + (size_t)(colBase + scol) * K + skp;
    // low 32 bits of a generic LDS pointer == LDS byte address (ISA 10.2)
    const uint32_t ldsOff =
        (uint32_t)(uintptr_t)&sB[scol * 64 + skp];

    const float*  ArowF = (const float*)Av  + (size_t)(tileM * 16 + lm) * K;
    const __bf16* ArowH = (const __bf16*)Av + (size_t)(tileM * 16 + lm) * K;

    v8f acc[4];
#pragma unroll
    for (int t = 0; t < 4; ++t)
        acc[t] = (v8f){0.f,0.f,0.f,0.f,0.f,0.f,0.f,0.f};

    for (int kc = 0; kc < K; kc += 64) {
        // ---- async stage of the 64x64 B chunk into LDS ----
        uint64_t ga = (uint64_t)(uintptr_t)(gB + kc);
        asm volatile("global_load_async_to_lds_b128 %0, %1, off"
                     :: "v"(ldsOff), "v"(ga) : "memory");
        asm volatile("global_load_async_to_lds_b128 %0, %1, off offset:16"
                     :: "v"(ldsOff), "v"(ga) : "memory");
        asm volatile("s_wait_asynccnt 0" ::: "memory");
        __syncthreads();

#pragma unroll
        for (int ks = 0; ks < 64; ks += 32) {
            // ---- A fragment at K = kc+ks ----
            v16bf a;
            if (A_BF16) {
                const __bf16* ap = ArowH + kc + ks;
                v8bf a0 = *(const v8bf*)(ap + kA0);
                v8bf a1 = *(const v8bf*)(ap + kA1);
#pragma unroll
                for (int e = 0; e < 8; ++e) { a[e] = a0[e]; a[8 + e] = a1[e]; }
            } else {
                const float* ap = ArowF + kc + ks;
                float4 f0 = *(const float4*)(ap + kA0);
                float4 f1 = *(const float4*)(ap + kA0 + 4);
                float4 f2 = *(const float4*)(ap + kA1);
                float4 f3 = *(const float4*)(ap + kA1 + 4);
                a[0]  = (__bf16)f0.x; a[1]  = (__bf16)f0.y;
                a[2]  = (__bf16)f0.z; a[3]  = (__bf16)f0.w;
                a[4]  = (__bf16)f1.x; a[5]  = (__bf16)f1.y;
                a[6]  = (__bf16)f1.z; a[7]  = (__bf16)f1.w;
                a[8]  = (__bf16)f2.x; a[9]  = (__bf16)f2.y;
                a[10] = (__bf16)f2.z; a[11] = (__bf16)f2.w;
                a[12] = (__bf16)f3.x; a[13] = (__bf16)f3.y;
                a[14] = (__bf16)f3.z; a[15] = (__bf16)f3.w;
            }
#pragma unroll
            for (int t = 0; t < 4; ++t) {
                const __bf16* bp = &sB[(t * 16 + lm) * 64 + ks + kB];
                v16bf b = *(const v16bf*)bp;      // 2x ds_load_b128
                acc[t] = __builtin_amdgcn_wmma_f32_16x16x32_bf16(
                    false, a, false, b, (short)0, acc[t], false, false);
            }
        }
        __syncthreads();
    }

    // ---- epilogue (fully branch-free per instantiation) ----
#pragma unroll
    for (int t = 0; t < 4; ++t) {
        int col = colBase + t * 16 + lm;
        float bv = bias[col];
#pragma unroll
        for (int r = 0; r < 8; ++r) {
            int row = tileM * 16 + hi * 8 + r;
            float v = acc[t][r] + bv;
            if (RELU) v = v > 0.f ? v : 0.f;
            if (MASK) { if (rowMask[row]) v = 0.f; }
            size_t o = (size_t)row * Nc + col;
            if (OUT_BF16) ((__bf16*)outv)[o] = (__bf16)v;
            else          ((float*)outv)[o] = v;
        }
    }
}

// ---------------------------------------------------------------------------
// MS-deformable 3D sampling. One wave per (n, q, head); lane = head channel
// (HD=32) so each corner gather is a coalesced 64B line from the bf16 value
// tensor [N, LIN, NH, HD]. Softmax over the 16 (level,point) logits is
// recomputed branch-free (no private arrays -> no scratch).
// ---------------------------------------------------------------------------
__global__ __launch_bounds__(256)
void deform_sample_kernel(const float* __restrict__ ref,     // [N,LQ,NL,3]
                          const float* __restrict__ off,     // [N,LQ,384]
                          const float* __restrict__ aw,      // [N,LQ,128]
                          const __bf16* __restrict__ value,  // [N,LIN,NH,HD]
                          const int* __restrict__ shapes,    // [NL,3]
                          const int* __restrict__ lsi,       // [NL]
                          float* __restrict__ outAcc)        // [N,LQ,NH,HD]
{
    const int wid  = threadIdx.x >> 5;
    const int lane = threadIdx.x & 31;
    const int gw = blockIdx.x * 8 + wid;       // (n*LQ + q)*NH + h
    const int h  = gw & (NH_ - 1);
    const int nq = gw >> 3;
    const int n  = nq >> 10;
    if (n >= N_BATCH) return;

    const float* awp = aw + (size_t)nq * (NH_ * NL_ * NP_) + h * 16;
    float mx = -3.0e38f;
    for (int i = 0; i < 16; ++i) mx = fmaxf(mx, awp[i]);
    float ssum = 0.f;
    for (int i = 0; i < 16; ++i) ssum += __expf(awp[i] - mx);
    const float sinv = 1.f / ssum;

    const float*  refb = ref + (size_t)nq * (NL_ * 3);
    const float*  offb = off + (size_t)nq * (NH_ * NL_ * NP_ * 3) + h * 48;
    const __bf16* vb   = value + (size_t)n * LIN_ * (NH_ * HD_)
                               + h * HD_ + lane;

    float acc = 0.f;
    for (int l = 0; l < NL_; ++l) {
        const int Tl = shapes[l * 3 + 0];
        const int Hl = shapes[l * 3 + 1];
        const int Wl = shapes[l * 3 + 2];
        const float fT = (float)Tl, fH = (float)Hl, fW = (float)Wl;
        const int start = lsi[l];
        const float rt = refb[l * 3 + 0];
        const float rh = refb[l * 3 + 1];
        const float rw = refb[l * 3 + 2];
        for (int p = 0; p < NP_; ++p) {
            const float* o3 = offb + (l * 4 + p) * 3;
            // loc*sz - 0.5 with loc = ref + off/sz  ==  ref*sz + off - 0.5
            float pt = rt * fT + o3[0] - 0.5f;
            float ph = rh * fH + o3[1] - 0.5f;
            float pw = rw * fW + o3[2] - 0.5f;
            float t0f = floorf(pt), h0f = floorf(ph), w0f = floorf(pw);
            float ft = pt - t0f, fh = ph - h0f, fw = pw - w0f;
            int t0 = (int)t0f, h0 = (int)h0f, w0 = (int)w0f;

            float samp = 0.f;
#pragma unroll
            for (int dt = 0; dt < 2; ++dt)
#pragma unroll
            for (int dy = 0; dy < 2; ++dy)
#pragma unroll
            for (int dx = 0; dx < 2; ++dx) {
                int it = t0 + dt, iy = h0 + dy, ix = w0 + dx;
                float w = (dt ? ft : 1.f - ft) *
                          (dy ? fh : 1.f - fh) *
                          (dx ? fw : 1.f - fw);
                bool valid = (it >= 0) & (it < Tl) & (iy >= 0) & (iy < Hl) &
                             (ix >= 0) & (ix < Wl);
                int ct = min(max(it, 0), Tl - 1);
                int cy = min(max(iy, 0), Hl - 1);
                int cx = min(max(ix, 0), Wl - 1);
                int idx = start + (ct * Hl + cy) * Wl + cx;
                float val = (float)vb[(size_t)idx * (NH_ * HD_)];
                samp += valid ? (w * val) : 0.f;
            }
            acc += __expf(awp[l * 4 + p] - mx) * sinv * samp;
        }
    }
    outAcc[(size_t)nq * (NH_ * HD_) + h * HD_ + lane] = acc;
}

// ---------------------------------------------------------------------------
// Residual + LayerNorm over last dim (256). One wave per row; each lane owns
// 8 contiguous channels; shfl-xor reduction (wave32).
// ---------------------------------------------------------------------------
__global__ __launch_bounds__(256)
void ln_residual_kernel(const float* __restrict__ A, const float* __restrict__ B,
                        const float* __restrict__ g, const float* __restrict__ bb,
                        float* __restrict__ out, int rows)
{
    const int wid  = threadIdx.x >> 5;
    const int lane = threadIdx.x & 31;
    const int row = blockIdx.x * 8 + wid;
    if (row >= rows) return;

    const float* pa = A + (size_t)row * D_MODEL + lane * 8;
    const float* pb = B + (size_t)row * D_MODEL + lane * 8;
    float x[8];
    float s = 0.f;
#pragma unroll
    for (int i = 0; i < 8; ++i) { x[i] = pa[i] + pb[i]; s += x[i]; }
#pragma unroll
    for (int m = 16; m >= 1; m >>= 1) s += __shfl_xor(s, m, 32);
    const float mean = s * (1.f / D_MODEL);

    float ss = 0.f;
#pragma unroll
    for (int i = 0; i < 8; ++i) { float d = x[i] - mean; ss += d * d; }
#pragma unroll
    for (int m = 16; m >= 1; m >>= 1) ss += __shfl_xor(ss, m, 32);
    const float rstd = rsqrtf(ss * (1.f / D_MODEL) + 1e-5f);

    float* po = out + (size_t)row * D_MODEL + lane * 8;
#pragma unroll
    for (int i = 0; i < 8; ++i) {
        int col = lane * 8 + i;
        po[i] = (x[i] - mean) * rstd * g[col] + bb[col];
    }
}

// ---------------------------------------------------------------------------
// Host orchestration
// ---------------------------------------------------------------------------
extern "C" void kernel_launch(void* const* d_in, const int* in_sizes, int n_in,
                              void* d_out, int out_size, void* d_ws, size_t ws_size,
                              hipStream_t stream)
{
    (void)in_sizes; (void)n_in; (void)out_size; (void)ws_size;

    const float* tgt   = (const float*)d_in[0];   // [8,1024,256]
    const float* qpos  = (const float*)d_in[1];   // [8,1024,256]
    const float* refp  = (const float*)d_in[2];   // [8,1024,4,3]
    const float* src   = (const float*)d_in[3];   // [8,43520,256]
    const int*   shp   = (const int*)  d_in[4];   // [4,3]
    const int*   lsi   = (const int*)  d_in[5];   // [4]
    const unsigned char* pmask = (const unsigned char*)d_in[6]; // [8,43520]
    const float* vpw   = (const float*)d_in[7];
    const float* vpb   = (const float*)d_in[8];
    const float* sow   = (const float*)d_in[9];
    const float* sob   = (const float*)d_in[10];
    const float* aww   = (const float*)d_in[11];
    const float* awb   = (const float*)d_in[12];
    const float* opw   = (const float*)d_in[13];
    const float* opb   = (const float*)d_in[14];
    const float* ln1g  = (const float*)d_in[15];
    const float* ln1b  = (const float*)d_in[16];
    const float* l1w   = (const float*)d_in[17];
    const float* l1b   = (const float*)d_in[18];
    const float* l2w   = (const float*)d_in[19];
    const float* l2b   = (const float*)d_in[20];
    const float* ln3g  = (const float*)d_in[21];
    const float* ln3b  = (const float*)d_in[22];
    float* out = (float*)d_out;

    const int MQ = N_BATCH * LQ_;     // 8192
    const int MV = N_BATCH * LIN_;    // 348160

    // --- workspace carve ---
    char* base = (char*)d_ws;
    size_t off = 0;
    auto alloc = [&](size_t bytes) -> char* {
        char* p = base + off;
        off += (bytes + 255) & ~(size_t)255;
        return p;
    };
    __bf16* wtv  = (__bf16*)alloc((size_t)D_MODEL * D_MODEL * 2);
    __bf16* wto  = (__bf16*)alloc((size_t)D_MODEL * 384 * 2);
    __bf16* wta  = (__bf16*)alloc((size_t)D_MODEL * 128 * 2);
    __bf16* wtp  = (__bf16*)alloc((size_t)D_MODEL * D_MODEL * 2);
    __bf16* wt1  = (__bf16*)alloc((size_t)D_MODEL * DFF_ * 2);
    __bf16* wt2  = (__bf16*)alloc((size_t)DFF_ * D_MODEL * 2);
    float*  qb   = (float*)alloc((size_t)MQ * D_MODEL * 4);
    float*  offb = (float*)alloc((size_t)MQ * 384 * 4);
    float*  awbf = (float*)alloc((size_t)MQ * 128 * 4);
    float*  atto = (float*)alloc((size_t)MQ * D_MODEL * 4);
    float*  tgt2 = (float*)alloc((size_t)MQ * D_MODEL * 4);
    float*  xbuf = (float*)alloc((size_t)MQ * D_MODEL * 4);
    __bf16* hff  = (__bf16*)alloc((size_t)MQ * DFF_ * 2);
    float*  f2   = (float*)alloc((size_t)MQ * D_MODEL * 4);
    __bf16* valb = (__bf16*)alloc((size_t)MV * D_MODEL * 2);

    // --- weight convert/transpose (tiny, L2 resident) ---
    auto wcvt = [&](const float* W, __bf16* WT, int K, int Nc) {
        int n = K * Nc;
        wcvt_kernel<<<(n + 255) / 256, 256, 0, stream>>>(W, WT, K, Nc);
    };
    wcvt(vpw, wtv, D_MODEL, D_MODEL);
    wcvt(sow, wto, D_MODEL, 384);
    wcvt(aww, wta, D_MODEL, 128);
    wcvt(opw, wtp, D_MODEL, D_MODEL);
    wcvt(l1w, wt1, D_MODEL, DFF_);
    wcvt(l2w, wt2, DFF_, D_MODEL);

    // --- q = tgt + query_pos ---
    {
        int n = MQ * D_MODEL;
        addvec_kernel<<<(n + 255) / 256, 256, 0, stream>>>(tgt, qpos, qb, n);
    }

    auto blocks_for = [](int M, int Nc) { return (M / 128) * (Nc / 64); };

    // --- value projection (the big one): bf16 out + pad mask ---
    gemm_wmma_kernel<false, true, false, true>
        <<<blocks_for(MV, D_MODEL), 256, 0, stream>>>(
            src, wtv, vpb, valb, pmask, MV, D_MODEL, D_MODEL);
    // --- sampling offsets and attention logits from q ---
    gemm_wmma_kernel<false, false, false, false>
        <<<blocks_for(MQ, 384), 256, 0, stream>>>(
            qb, wto, sob, offb, nullptr, MQ, 384, D_MODEL);
    gemm_wmma_kernel<false, false, false, false>
        <<<blocks_for(MQ, 128), 256, 0, stream>>>(
            qb, wta, awb, awbf, nullptr, MQ, 128, D_MODEL);

    // --- deformable trilinear sampling ---
    {
        int waves = MQ * NH_;                       // 65536
        deform_sample_kernel<<<(waves + 7) / 8, 256, 0, stream>>>(
            refp, offb, awbf, valb, shp, lsi, atto);
    }

    // --- output projection ---
    gemm_wmma_kernel<false, false, false, false>
        <<<blocks_for(MQ, D_MODEL), 256, 0, stream>>>(
            atto, wtp, opb, tgt2, nullptr, MQ, D_MODEL, D_MODEL);

    // --- x = LN1(tgt + tgt2) ---
    ln_residual_kernel<<<(MQ + 7) / 8, 256, 0, stream>>>(
        tgt, tgt2, ln1g, ln1b, xbuf, MQ);

    // --- FFN: lin1 (ReLU, bf16 out) then lin2 (bf16 A) ---
    gemm_wmma_kernel<false, true, true, false>
        <<<blocks_for(MQ, DFF_), 256, 0, stream>>>(
            xbuf, wt1, l1b, hff, nullptr, MQ, DFF_, D_MODEL);
    gemm_wmma_kernel<true, false, false, false>
        <<<blocks_for(MQ, D_MODEL), 256, 0, stream>>>(
            hff, wt2, l2b, f2, nullptr, MQ, D_MODEL, DFF_);

    // --- out = LN3(x + ffn) ---
    ln_residual_kernel<<<(MQ + 7) / 8, 256, 0, stream>>>(
        xbuf, f2, ln3g, ln3b, out, MQ);
}
